// StochasticMLP_76673756168569
// MI455X (gfx1250) — matrix operand order repaired and checked by reading, not compiled, not run on hardware
//
#include <hip/hip_runtime.h>

// ---------------------------------------------------------------------------
// StochasticMLP Gibbs sampler for MI455X (gfx1250, wave32).
//   - 6 GEMMs via v_wmma_f32_16x16x32_bf16, fully specialized by template so
//     the hot K-loop has zero branches / zero guarded loads (b128 loads for A and transposed B).
//   - K-tails and the N=10 case use clamped in-bounds addresses + select-0
//     (v_cndmask) instead of conditional loads.
//   - layer-1 Gibbs scan: one wave per batch row, 8 child logits per lane,
//     wave32 shuffle reduction, global_prefetch_b8 of the next W2 row.
//   - layer-2 Gibbs scan: one thread per batch row (10 children in registers).
// ---------------------------------------------------------------------------

typedef __attribute__((ext_vector_type(16))) __bf16 v16bf;
typedef __attribute__((ext_vector_type(8)))  float  v8f;

enum : int { NB = 2048, DIN = 784, NH = 256, NOUT = 10, OUTW = NH + NH + NOUT };

static __device__ __forceinline__ __bf16 f2bf(float f) {
  // round-to-nearest-even f32 -> bf16
  unsigned u = __builtin_bit_cast(unsigned, f);
  u += 0x7FFFu + ((u >> 16) & 1u);
  unsigned short h = (unsigned short)(u >> 16);
  return __builtin_bit_cast(__bf16, h);
}

static __device__ __forceinline__ float softplus_f(float x) {
  float a = fabsf(x);
  return fmaxf(x, 0.0f) + __logf(1.0f + __expf(-a));
}
static __device__ __forceinline__ float sigmoid_f(float x) {
  return 1.0f / (1.0f + __expf(-x));
}

// ---------------------------------------------------------------------------
// C[2048,NMAT] = cvt_bf16(A[2048,KMAT]) x cvt_bf16(B) (+bias), f32 accum.
// TRANSB=0: B[k][n] = Bw[k*LDB + n];  TRANSB=1: B[k][n] = Bw[n*LDB + k].
// One 16x16 tile per wave; 8 waves per block; grids sized exactly, so EXEC
// is all-ones around every WMMA.
// ---------------------------------------------------------------------------
template<bool AISINT, bool TRANSB, int LDA, int LDB, int NMAT, int KMAT, bool HASBIAS>
__global__ __launch_bounds__(256) void wmma_gemm(
    const void* __restrict__ Av, const float* __restrict__ Bw,
    const float* __restrict__ bias, float* __restrict__ C)
{
  constexpr int TILESM = NB / 16;          // 128
  const int lane = threadIdx.x & 31;
  const int wave = threadIdx.x >> 5;
  const int tile = blockIdx.x * 8 + wave;
  const int tm   = tile % TILESM;
  const int tn   = tile / TILESM;
  const int half = lane >> 4;              // 0: lanes 0-15, 1: lanes 16-31
  const int lm   = lane & 15;
  const int m    = tm * 16 + lm;
  const int ncol = tn * 16 + lm;
  // clamped (always in-bounds) column + validity flag; constant-folds to
  // identity/true when NMAT is a multiple of 16.
  const int  ncl = (NMAT % 16) ? ((ncol < NMAT) ? ncol : (NMAT - 1)) : ncol;
  const bool nok = (NMAT % 16) ? (ncol < NMAT) : true;

  const float* Af = (const float*)Av;
  const int*   Ai = (const int*)Av;

  v8f acc = {};
  constexpr int FULL = KMAT / 32;

  // ---- main K loop: no bounds checks, b128 loads for A and transposed B --------------------
  for (int kt = 0; kt < FULL; ++kt) {
    const int kb = kt * 32;
    v16bf a, b;

    // A 16x32 bf16 fragment: lane m holds 16 consecutive k from kb+half*16.
    #pragma unroll
    for (int g = 0; g < 2; ++g) {
      const int k0 = kb + half * 16 + g * 8;
      if constexpr (AISINT) {
        const int4* p = (const int4*)(Ai + m * LDA + k0);
        int4 r0 = p[0], r1 = p[1];
        const int rv[8] = {r0.x, r0.y, r0.z, r0.w, r1.x, r1.y, r1.z, r1.w};
        #pragma unroll
        for (int i = 0; i < 8; ++i) a[g * 8 + i] = f2bf((float)rv[i]);
      } else {
        const float4* p = (const float4*)(Af + m * LDA + k0);
        float4 r0 = p[0], r1 = p[1];
        const float rv[8] = {r0.x, r0.y, r0.z, r0.w, r1.x, r1.y, r1.z, r1.w};
        #pragma unroll
        for (int i = 0; i < 8; ++i) a[g * 8 + i] = f2bf(rv[i]);
      }
    }
    // B 32x16 bf16 fragment: lane n holds 16 consecutive k from kb+half*16.
    {
      const int k0 = kb + half * 16;
      if constexpr (TRANSB) {
        const float4* p = (const float4*)(Bw + ncl * LDB + k0);
        #pragma unroll
        for (int q = 0; q < 4; ++q) {
          float4 r = p[q];
          const float rv[4] = {r.x, r.y, r.z, r.w};
          #pragma unroll
          for (int i = 0; i < 4; ++i) {
            float v = rv[i];
            if constexpr (NMAT % 16) v = nok ? v : 0.0f;
            b[q * 4 + i] = f2bf(v);
          }
        }
      } else {
        #pragma unroll
        for (int i = 0; i < 16; ++i) {
          float v = Bw[(k0 + i) * LDB + ncl];
          if constexpr (NMAT % 16) v = nok ? v : 0.0f;
          b[i] = f2bf(v);
        }
      }
    }
    acc = __builtin_amdgcn_wmma_f32_16x16x32_bf16(
        false, a, false, b, (short)0, acc, false, false);
  }

  // ---- peeled K tail: clamped addresses + select-0, unconditional loads ----
  if constexpr (KMAT % 32 != 0) {
    const int kb = FULL * 32;
    v16bf a, b;
    #pragma unroll
    for (int g = 0; g < 2; ++g) {
      const int k0 = kb + half * 16 + g * 8;
      #pragma unroll
      for (int i = 0; i < 8; ++i) {
        const int k  = k0 + i;
        const int kc = (k < KMAT) ? k : (KMAT - 1);
        float v;
        if constexpr (AISINT) v = (float)Ai[m * LDA + kc];
        else                  v = Af[m * LDA + kc];
        a[g * 8 + i] = f2bf((k < KMAT) ? v : 0.0f);
      }
    }
    {
      const int k0 = kb + half * 16;
      #pragma unroll
      for (int i = 0; i < 16; ++i) {
        const int k  = k0 + i;
        const int kc = (k < KMAT) ? k : (KMAT - 1);
        float v;
        if constexpr (TRANSB) v = Bw[ncl * LDB + kc];
        else                  v = Bw[kc * LDB + ncl];
        b[i] = f2bf((k < KMAT && nok) ? v : 0.0f);
      }
    }
    acc = __builtin_amdgcn_wmma_f32_16x16x32_bf16(
        false, a, false, b, (short)0, acc, false, false);
  }

  float bv = 0.0f;
  if constexpr (HASBIAS) bv = nok ? bias[ncl] : 0.0f;
  #pragma unroll
  for (int r = 0; r < 8; ++r) {
    const int mm = tm * 16 + r + half * 8;   // C layout: VGPR r <-> M=r (+8 upper half)
    if (nok) C[mm * NMAT + ncol] = acc[r] + bv;
  }
}

// ---------------------------------------------------------------------------
// Layer-1 Gibbs scan: 256 children. One wave per batch row; lane holds
// children [8*lane, 8*lane+8). nl stays in registers for all 256 steps.
//   prob_j = sigmoid(pl_j + wdott_j - sum_k[sp(nl0_k+w_jk) - sp(nl0_k)])
// ---------------------------------------------------------------------------
__global__ __launch_bounds__(256) void gibbs_scan256(
    const float* __restrict__ nl0buf,  // [B,256] cv @ W2 + b2
    const float* __restrict__ pl,      // [B,256] parent logits
    const float* __restrict__ wdott,   // [B,256] nv @ W2^T
    const int*   __restrict__ cv,      // [B,256] old node values (h1)
    const float* __restrict__ W,       // [256,256] W2
    const float* __restrict__ u,       // [256,B] uniforms
    float* __restrict__ newbuf,        // [B,256] h1_new (contiguous, feeds pl2)
    float* __restrict__ dout)          // [B,OUTW] cols 0..255
{
  const int lane = threadIdx.x & 31;
  const int b    = blockIdx.x * 8 + (threadIdx.x >> 5);

  float nl[8];
  {
    const float4* p = (const float4*)(nl0buf + b * NH + lane * 8);
    float4 t0 = p[0], t1 = p[1];
    nl[0]=t0.x; nl[1]=t0.y; nl[2]=t0.z; nl[3]=t0.w;
    nl[4]=t1.x; nl[5]=t1.y; nl[6]=t1.z; nl[7]=t1.w;
  }

  for (int j = 0; j < NH; ++j) {
    const float4* wp = (const float4*)(W + j * NH + lane * 8);
    float4 w0 = wp[0], w1 = wp[1];
    if (j + 1 < NH)  // hide latency of next broadcast row (global_prefetch_b8)
      __builtin_prefetch(W + (j + 1) * NH + lane * 8, 0, 1);
    float w[8] = {w0.x, w0.y, w0.z, w0.w, w1.x, w1.y, w1.z, w1.w};

    const float node = (float)cv[b * NH + j];
    float s = 0.0f;
    #pragma unroll
    for (int i = 0; i < 8; ++i) {
      float n0 = nl[i] - node * w[i];
      s += softplus_f(n0 + w[i]) - softplus_f(n0);
      nl[i] = n0;
    }
    #pragma unroll
    for (int off = 16; off > 0; off >>= 1) s += __shfl_xor(s, off, 32);

    const float prob = sigmoid_f(pl[b * NH + j] + wdott[b * NH + j] - s);
    const float nv = (u[j * NB + b] < prob) ? 1.0f : 0.0f;
    #pragma unroll
    for (int i = 0; i < 8; ++i) nl[i] += nv * w[i];

    if (lane == 0) {
      newbuf[b * NH + j] = nv;
      dout[b * OUTW + j] = nv;
    }
  }
}

// ---------------------------------------------------------------------------
// Layer-2 Gibbs scan: 10 children. One thread per batch row; nl[10] in regs.
// Also emits final output-layer logits (cols 512..521).
// ---------------------------------------------------------------------------
__global__ __launch_bounds__(256) void gibbs_scan10(
    const float* __restrict__ nl0buf,  // [B,10] h2 @ Wout + bout
    const float* __restrict__ pl,      // [B,256] h1_new @ W2 + b2
    const float* __restrict__ wdott,   // [B,256] y @ Wout^T
    const int*   __restrict__ cv,      // [B,256] old h2
    const float* __restrict__ Wout,    // [256,10]
    const float* __restrict__ u,       // [256,B]
    float* __restrict__ dout)          // [B,OUTW]
{
  const int b = blockIdx.x * blockDim.x + threadIdx.x;

  float nl[NOUT];
  #pragma unroll
  for (int c = 0; c < NOUT; ++c) nl[c] = nl0buf[b * NOUT + c];

  for (int j = 0; j < NH; ++j) {
    const float node = (float)cv[b * NH + j];
    float w[NOUT];
    #pragma unroll
    for (int c = 0; c < NOUT; ++c) w[c] = Wout[j * NOUT + c];  // wave-uniform
    float s = 0.0f;
    #pragma unroll
    for (int c = 0; c < NOUT; ++c) {
      float n0 = nl[c] - node * w[c];
      s += softplus_f(n0 + w[c]) - softplus_f(n0);
      nl[c] = n0;
    }
    const float prob = sigmoid_f(pl[b * NH + j] + wdott[b * NH + j] - s);
    const float nv = (u[j * NB + b] < prob) ? 1.0f : 0.0f;
    #pragma unroll
    for (int c = 0; c < NOUT; ++c) nl[c] += nv * w[c];
    dout[b * OUTW + NH + j] = nv;
  }
  #pragma unroll
  for (int c = 0; c < NOUT; ++c) dout[b * OUTW + 2 * NH + c] = nl[c];
}

// ---------------------------------------------------------------------------
extern "C" void kernel_launch(void* const* d_in, const int* in_sizes, int n_in,
                              void* d_out, int out_size, void* d_ws, size_t ws_size,
                              hipStream_t stream) {
  (void)in_sizes; (void)n_in; (void)out_size; (void)ws_size;

  const float* x    = (const float*)d_in[0];
  const int*   h1   = (const int*)  d_in[1];
  const int*   h2   = (const int*)  d_in[2];
  const int*   y    = (const int*)  d_in[3];
  const float* W1   = (const float*)d_in[4];
  const float* b1   = (const float*)d_in[5];
  const float* W2   = (const float*)d_in[6];
  const float* b2   = (const float*)d_in[7];
  const float* Wout = (const float*)d_in[8];
  const float* bout = (const float*)d_in[9];
  const float* u1   = (const float*)d_in[10];
  const float* u2   = (const float*)d_in[11];
  float* out = (float*)d_out;

  float* ws   = (float*)d_ws;
  float* pl1  = ws;               // [B,256]
  float* nl10 = pl1  + NB * NH;   // [B,256]
  float* wt1  = nl10 + NB * NH;   // [B,256]
  float* h1n  = wt1  + NB * NH;   // [B,256]
  float* pl2  = h1n  + NB * NH;   // [B,256]
  float* wt2  = pl2  + NB * NH;   // [B,256]
  float* nl20 = wt2  + NB * NH;   // [B,10]

  const dim3 blk(256);
  const int blocksN256 = (128 * (NH / 16)) / 8;            // 2048 tiles / 8 waves
  const int blocksN16  = (128 * ((NOUT + 15) / 16)) / 8;   // 128 tiles / 8 waves

  // ---- layer 1 precompute -------------------------------------------------
  // pl1 = x @ W1 + b1
  wmma_gemm<false, false, DIN, NH, NH, DIN, true>
      <<<blocksN256, blk, 0, stream>>>(x, W1, b1, pl1);
  // nl1_0 = h1 @ W2 + b2
  wmma_gemm<true, false, NH, NH, NH, NH, true>
      <<<blocksN256, blk, 0, stream>>>(h1, W2, b2, nl10);
  // wdott1 = h2 @ W2^T
  wmma_gemm<true, true, NH, NH, NH, NH, false>
      <<<blocksN256, blk, 0, stream>>>(h2, W2, nullptr, wt1);

  // ---- layer 1 scan -------------------------------------------------------
  gibbs_scan256<<<NB / 8, blk, 0, stream>>>(nl10, pl1, wt1, h1, W2, u1, h1n, out);

  // ---- layer 2 precompute -------------------------------------------------
  // pl2 = h1_new @ W2 + b2
  wmma_gemm<false, false, NH, NH, NH, NH, true>
      <<<blocksN256, blk, 0, stream>>>(h1n, W2, b2, pl2);
  // nl2_0 = h2 @ Wout + bout   (N = 10)
  wmma_gemm<true, false, NH, NOUT, NOUT, NH, true>
      <<<blocksN16, blk, 0, stream>>>(h2, Wout, bout, nl20);
  // wdott2 = y @ Wout^T        (K = 10)
  wmma_gemm<true, true, NOUT, NOUT, NH, NOUT, false>
      <<<blocksN256, blk, 0, stream>>>(y, Wout, nullptr, wt2);

  // ---- layer 2 scan + final logits ---------------------------------------
  gibbs_scan10<<<NB / 256, blk, 0, stream>>>(nl20, pl2, wt2, h2, Wout, u2, out);
}
